// Char_Encoder_33062658244616
// MI455X (gfx1250) — compile-verified, matrix operand order checked by reading, not run on hardware
//
#include <hip/hip_runtime.h>
#include <hip/hip_bf16.h>

typedef _Float16 f16;
typedef __attribute__((ext_vector_type(16))) _Float16 v16h;
typedef __attribute__((ext_vector_type(8)))  _Float16 v8h;
typedef __attribute__((ext_vector_type(8)))  float    v8f;

#define NSEQ   16384     // 128*128 words
#define SEQ_T  32        // sequences per workgroup (2 WMMA M-tiles)
#define LSTEPS 16        // chars per word
#define CDIM   100       // char embedding dim
#define DPAD   128       // CDIM padded to 4 K-chunks of 32
#define HIDN   200       // hidden size
#define HPAD   224       // HIDN padded to 7 K-chunks of 32
#define KTOT   352       // DPAD + HPAD
#define GATES  800       // 4*HIDN
#define NTILE  50        // GATES/16 gate tiles
#define KCH    11        // 4 (x) + 7 (h) K-chunks of 32
#define TPB    320       // 10 waves * 32 (wave32)
#define TPW    5         // gate tiles per wave
#define WPK_HALFS (NTILE*KCH*32*16)   // 281600 halfs per direction

// ---- Pack fp32 weights into WMMA B-fragment layout (f16), K = [W_ih | pad | W_hh | pad]
// Fragment (nt,kc): 32 lanes x 16 halfs. Lane l: N = l&15. Element j:
//   K_local = (j>>3)*16 + (l>>4)*8 + (j&7)   (ISA 16-bit operand striping)
__global__ void pack_weights(const float* __restrict__ w_ih,
                             const float* __restrict__ w_hh,
                             f16* __restrict__ dst) {
    int tid = blockIdx.x * blockDim.x + threadIdx.x;   // 0 .. WPK_HALFS-1
    int j    = tid & 15;
    int lane = (tid >> 4) & 31;
    int kc   = (tid >> 9) % KCH;
    int nt   = tid / (KCH * 512);
    int n  = lane & 15;
    int hi = lane >> 4;
    int kk = ((j >> 3) << 4) + (hi << 3) + (j & 7);
    int g = nt * 16 + n;
    int k = kc * 32 + kk;
    float v = 0.f;
    if (k < DPAD) {
        if (k < CDIM) v = w_ih[g * CDIM + k];
    } else {
        int kh = k - DPAD;
        if (kh < HIDN) v = w_hh[g * HIDN + kh];
    }
    dst[tid] = (f16)v;
}

__global__ void pack_bias(const float* __restrict__ bi,
                          const float* __restrict__ bh,
                          float* __restrict__ dst) {
    int i = blockIdx.x * blockDim.x + threadIdx.x;
    if (i < GATES) dst[i] = bi[i] + bh[i];
}

// ---- Fused bidirectional char-LSTM. blockIdx.x = 32-sequence tile, blockIdx.y = direction.
__global__ __launch_bounds__(TPB, 1)   // 1 WG/CU: give RA the full VGPR budget (no spills)
void char_lstm(const int* __restrict__ char_ids, const float* __restrict__ char_emb,
               const f16* __restrict__ wpkF, const f16* __restrict__ wpkB,
               const float* __restrict__ biasF, const float* __restrict__ biasB,
               float* __restrict__ out) {
    // A operand: [x_t (128) | h (224, zero-padded past 200)] per sequence, f16
    __shared__ f16  abuf[SEQ_T][KTOT];      // 22.5 KB
    __shared__ f16  gatesS[16][GATES];      // 25.6 KB (one M-group at a time)
    __shared__ int  idsS[SEQ_T][LSTEPS];    //  2 KB
    __shared__ int  lensS[SEQ_T];

    const int tid = threadIdx.x;
    const int dir = blockIdx.y;
    const int blk = blockIdx.x;
    const f16*   wpk  = dir ? wpkB  : wpkF;
    const float* bias = dir ? biasB : biasF;

    // init: ids + lens (one thread per sequence); zero h region of abuf
    if (tid < SEQ_T) {
        const int* p = char_ids + (size_t)(blk * SEQ_T + tid) * LSTEPS;
        int cnt = 0;
        #pragma unroll
        for (int j = 0; j < LSTEPS; ++j) { int id = p[j]; idsS[tid][j] = id; cnt += (id != 0); }
        lensS[tid] = cnt > 0 ? cnt : 1;
    }
    for (int i = tid; i < SEQ_T * HPAD; i += TPB) {
        int s = i / HPAD, d = i % HPAD;
        abuf[s][DPAD + d] = (f16)0.f;
    }
    __syncthreads();

    const int wv = tid >> 5, lane = tid & 31;
    const int mrow = lane & 15, hih = lane >> 4;

    float bt[TPW];
    #pragma unroll
    for (int i = 0; i < TPW; ++i) bt[i] = bias[(wv * TPW + i) * 16 + mrow];

    // per-thread cell state: phase p -> sequence (p*16 + tid/20), 10 hidden units
    float hreg[2][10], creg[2][10];
    #pragma unroll
    for (int p = 0; p < 2; ++p)
        #pragma unroll
        for (int k = 0; k < 10; ++k) { hreg[p][k] = 0.f; creg[p][k] = 0.f; }
    const int pseq = tid / 20;
    const int phh0 = (tid * 10) % HIDN;

    // Opaque 32-bit fragment-index offset (SGPR), laundered every step:
    // blocks LICM from hoisting the loop-invariant B loads (round-1 spill storm)
    // while keeping the wpk pointer chain visible -> addrspace(1) global_load_b128.
    unsigned woff = 0;

    #pragma unroll 1
    for (int tt = 0; tt < LSTEPS; ++tt) {
        const int t = dir ? (LSTEPS - 1 - tt) : tt;
        asm volatile("" : "+s"(woff));

        // stage 1: gather char embedding for step t into abuf x-region (f16, zero-padded)
        for (int i = tid; i < SEQ_T * DPAD; i += TPB) {
            int s = i >> 7, d = i & 127;
            float v = 0.f;
            if (d < CDIM) v = char_emb[(size_t)idsS[s][t] * CDIM + d];
            abuf[s][d] = (f16)v;
        }
        __syncthreads();

        // stage 2: gates(32x800) = [x_t | h] @ W^T + bias  via v_wmma_f32_16x16x32_f16
        // Each B fragment feeds TWO WMMAs (M-groups 0 and 1): in-register weight reuse.
        v8f acc[2][TPW];
        #pragma unroll
        for (int p = 0; p < 2; ++p)
            #pragma unroll
            for (int i = 0; i < TPW; ++i) {
                v8f a;
                #pragma unroll
                for (int r = 0; r < 8; ++r) a[r] = bt[i];
                acc[p][i] = a;
            }
        #pragma unroll
        for (int kc = 0; kc < KCH; ++kc) {
            const int k0 = kc * 32;
            v16h av[2];
            #pragma unroll
            for (int p = 0; p < 2; ++p) {
                const f16* ap = &abuf[0][0] + (p * 16 + mrow) * KTOT + k0 + hih * 8;
                v8h alo = *(const v8h*)ap;          // K = base .. base+7   (ds_load_b128)
                v8h ahi = *(const v8h*)(ap + 16);   // K = base+16 .. +23
                #pragma unroll
                for (int r = 0; r < 8; ++r) { av[p][r] = alo[r]; av[p][r + 8] = ahi[r]; }
            }
            #pragma unroll
            for (int i = 0; i < TPW; ++i) {
                int nt = wv * TPW + i;
                v16h bv = *((const v16h*)wpk +
                            ((size_t)(woff + (unsigned)((nt * KCH + kc) * 32 + lane))));
                acc[0][i] = __builtin_amdgcn_wmma_f32_16x16x32_f16(
                    false, av[0], false, bv, (short)0, acc[0][i], false, false);
                acc[1][i] = __builtin_amdgcn_wmma_f32_16x16x32_f16(
                    false, av[1], false, bv, (short)0, acc[1][i], false, false);
            }
        }

        // stages 3a/3b: per M-group, round-trip gates through LDS, pointwise LSTM cell
        #pragma unroll 1
        for (int p = 0; p < 2; ++p) {
            // scatter D tiles (lanes 0-15 -> M=r, lanes 16-31 -> M=8+r)
            #pragma unroll
            for (int i = 0; i < TPW; ++i) {
                int nt = wv * TPW + i;
                #pragma unroll
                for (int r = 0; r < 8; ++r)
                    gatesS[hih * 8 + r][nt * 16 + mrow] = (f16)(p ? acc[1][i][r]
                                                                  : acc[0][i][r]);
            }
            __syncthreads();

            const int sl = p * 16 + pseq;            // sequence within workgroup
            const bool valid = t < lensS[sl];
            #pragma unroll
            for (int k = 0; k < 10; ++k) {
                int hh = phh0 + k;
                float gi = (float)gatesS[pseq][hh];
                float gf = (float)gatesS[pseq][HIDN + hh];
                float gg = (float)gatesS[pseq][2 * HIDN + hh];
                float go = (float)gatesS[pseq][3 * HIDN + hh];
                float si = 1.f / (1.f + __expf(-gi));
                float sf = 1.f / (1.f + __expf(-gf));
                float so = 1.f / (1.f + __expf(-go));
                float tg = 1.f - 2.f / (__expf(2.f * gg) + 1.f);   // tanh
                float cn = sf * creg[p][k] + si * tg;
                float tc = 1.f - 2.f / (__expf(2.f * cn) + 1.f);   // tanh
                float hn = so * tc;
                if (valid) { creg[p][k] = cn; hreg[p][k] = hn; }
                abuf[sl][DPAD + hh] = (f16)hreg[p][k];   // next step's A operand
            }
            __syncthreads();
        }
    }

    // final hidden -> out[..., dir*200 : dir*200+200]  (fp32)
    #pragma unroll
    for (int p = 0; p < 2; ++p) {
        float* op = out + ((size_t)(blk * SEQ_T + p * 16 + pseq)) * (2 * HIDN)
                        + dir * HIDN + phh0;
        #pragma unroll
        for (int k = 0; k < 10; ++k) op[k] = hreg[p][k];
    }
}

extern "C" void kernel_launch(void* const* d_in, const int* in_sizes, int n_in,
                              void* d_out, int out_size, void* d_ws, size_t ws_size,
                              hipStream_t stream) {
    const int*   char_ids = (const int*)  d_in[0];
    const float* char_emb = (const float*)d_in[1];
    const float* w_ih_f = (const float*)d_in[2];
    const float* w_hh_f = (const float*)d_in[3];
    const float* b_ih_f = (const float*)d_in[4];
    const float* b_hh_f = (const float*)d_in[5];
    const float* w_ih_b = (const float*)d_in[6];
    const float* w_hh_b = (const float*)d_in[7];
    const float* b_ih_b = (const float*)d_in[8];
    const float* b_hh_b = (const float*)d_in[9];
    float* out = (float*)d_out;

    f16*   wpkF  = (f16*)d_ws;
    f16*   wpkB  = wpkF + WPK_HALFS;
    float* biasF = (float*)(wpkB + WPK_HALFS);
    float* biasB = biasF + GATES;

    pack_weights<<<WPK_HALFS / 256, 256, 0, stream>>>(w_ih_f, w_hh_f, wpkF);
    pack_weights<<<WPK_HALFS / 256, 256, 0, stream>>>(w_ih_b, w_hh_b, wpkB);
    pack_bias<<<(GATES + 255) / 256, 256, 0, stream>>>(b_ih_f, b_hh_f, biasF);
    pack_bias<<<(GATES + 255) / 256, 256, 0, stream>>>(b_ih_b, b_hh_b, biasB);

    dim3 grid(NSEQ / SEQ_T, 2);
    char_lstm<<<grid, TPB, 0, stream>>>(char_ids, char_emb, wpkF, wpkB, biasF, biasB, out);
}
